// KANLayer_5686536700338
// MI455X (gfx1250) — compile-verified, matrix operand order checked by reading, not run on hardware
//
#include <hip/hip_runtime.h>

typedef __attribute__((ext_vector_type(16))) _Float16 v16h;
typedef __attribute__((ext_vector_type(8)))  _Float16 v8h;
typedef __attribute__((ext_vector_type(8)))  float    v8f;

#define IN_DIM  64
#define OUT_DIM 64
#define BATCH   2048
#define KORD    3
#define NBASIS  8              // NUM + K
#define FEAT    9              // 8 spline basis + silu
#define KDIM    (IN_DIM*FEAT)  // 576 = 18 * 32
#define KPAD    592            // LDS row stride in halfs (16B aligned, padded)

// ---------------------------------------------------------------------------
// Kernel 1: fold mask/scales/coef into a (64 x 576) f16 weight matrix
//   W[n][i*9+m] = mask[j]*scale_sp[j]*coef[j,m]   (m<8),  j = n*64+i
//   W[n][i*9+8] = mask[j]*scale_base[j]
// ---------------------------------------------------------------------------
__global__ void kan_prep_w(const float* __restrict__ coef,
                           const float* __restrict__ scale_base,
                           const float* __restrict__ scale_sp,
                           const float* __restrict__ maskp,
                           _Float16* __restrict__ W)
{
    int t = blockIdx.x * blockDim.x + threadIdx.x;
    if (t >= OUT_DIM * KDIM) return;
    int n = t / KDIM;
    int k = t - n * KDIM;
    int i = k / FEAT;
    int m = k - i * FEAT;
    int j = n * IN_DIM + i;
    float v = (m < NBASIS) ? maskp[j] * scale_sp[j] * coef[j * NBASIS + m]
                           : maskp[j] * scale_base[j];
    W[t] = (_Float16)v;
}

// ---------------------------------------------------------------------------
// Kernel 2: fused spline-feature build (LDS) + WMMA GEMM
//   block = 128 threads (4 waves, wave32), one block per 16 batch rows
//   out(2048x64) = F(2048x576) * W^T(576x64), f32 accumulate
// ---------------------------------------------------------------------------
__global__ void __launch_bounds__(128)
kan_fused(const float* __restrict__ x_in,     // (2048, 64)
          const float* __restrict__ grid,     // first row (6 pts) used
          const _Float16* __restrict__ W,     // (64, 576) f16 row-major
          float* __restrict__ out)            // (2048, 64)
{
    __shared__ _Float16 Fs[16 * KPAD];        // ~18.5 KB feature tile

    const int mtile = blockIdx.x;             // 0..127
    const int t     = threadIdx.x;            // 0..127

    // ---- extended uniform grid: 12 knots, spacing h ----
    float g0 = grid[0], g5 = grid[5];
    float h  = (g5 - g0) * 0.2f;
    float ge[12];
#pragma unroll
    for (int j = 0; j < 12; ++j)
        ge[j] = g0 + h * (float)(j - 3);      // linspace rows: grid[c]=g0+h*c

    // ---- Phase 1: per-x cubic B-spline basis + silu -> LDS ----
    const int r    = t & 15;                  // local batch row 0..15
    const int i0   = (t >> 4) * 8;            // 8 consecutive i per thread
    const int brow = mtile * 16 + r;
#pragma unroll
    for (int q = 0; q < 8; ++q) {
        int   i = i0 + q;
        float x = x_in[brow * IN_DIM + i];

        float B[11];
#pragma unroll
        for (int j = 0; j < 11; ++j)
            B[j] = (x >= ge[j] && x < ge[j + 1]) ? 1.0f : 0.0f;
#pragma unroll
        for (int p = 1; p <= KORD; ++p) {
#pragma unroll
            for (int j = 0; j < 11; ++j) {
                if (j + p < 11) {
                    float a = (x - ge[j])         / (ge[j + p]     - ge[j]);
                    float b = (ge[j + p + 1] - x) / (ge[j + p + 1] - ge[j + 1]);
                    B[j] = a * B[j] + b * B[j + 1];
                }
            }
        }
        float silu = x / (1.0f + __expf(-x));

        _Float16* dst = &Fs[r * KPAD + i * FEAT];
#pragma unroll
        for (int m = 0; m < NBASIS; ++m) dst[m] = (_Float16)B[m];
        dst[NBASIS] = (_Float16)silu;
    }
    __syncthreads();

    // ---- Phase 2: WMMA over K=576 (18 steps of 32) ----
    const int lane   = t & 31;
    const int wv     = t >> 5;                 // N-tile 0..3
    const int lane16 = lane & 15;
    const int hi     = lane >> 4;              // 0: K 0-7/16-23, 1: K 8-15/24-31

    union Frag { v16h v; v8h h[2]; };
    const _Float16* frow = Fs + lane16 * KPAD;           // A: M = lane%16
    const _Float16* wrow = W  + (wv * 16 + lane16) * KDIM; // B: N = lane%16

    v8f acc = {};
#pragma unroll
    for (int s = 0; s < KDIM / 32; ++s) {
        int k0 = s * 32 + hi * 8;
        Frag a, b;
        a.h[0] = *(const v8h*)(frow + k0);
        a.h[1] = *(const v8h*)(frow + k0 + 16);
        b.h[0] = *(const v8h*)(wrow + k0);
        b.h[1] = *(const v8h*)(wrow + k0 + 16);
        acc = __builtin_amdgcn_wmma_f32_16x16x32_f16(
            /*neg_a=*/false, a.v, /*neg_b=*/false, b.v,
            /*c_mod=*/(short)0, acc, /*reuse_a=*/false, /*reuse_b=*/false);
    }

    // ---- Store D: VGPR r -> M = r + 8*(lane>=16), N = lane%16 ----
    const int colOut  = wv * 16 + lane16;
    const int rowBase = mtile * 16 + (hi << 3);
#pragma unroll
    for (int rr = 0; rr < 8; ++rr)
        out[(rowBase + rr) * OUT_DIM + colOut] = acc[rr];
}

// ---------------------------------------------------------------------------
extern "C" void kernel_launch(void* const* d_in, const int* in_sizes, int n_in,
                              void* d_out, int out_size, void* d_ws, size_t ws_size,
                              hipStream_t stream)
{
    const float* inputs = (const float*)d_in[0];  // (2048, 64)
    const float* grid   = (const float*)d_in[1];  // (4096, 6) identical rows
    const float* coef   = (const float*)d_in[2];  // (4096, 8)
    const float* sbase  = (const float*)d_in[3];  // (4096,)
    const float* ssp    = (const float*)d_in[4];  // (4096,)
    const float* maskp  = (const float*)d_in[5];  // (4096,)
    float*       outp   = (float*)d_out;          // (2048, 64)
    _Float16*    W      = (_Float16*)d_ws;        // 64*576*2 = 72 KB scratch

    kan_prep_w<<<(OUT_DIM * KDIM + 255) / 256, 256, 0, stream>>>(
        coef, sbase, ssp, maskp, W);

    kan_fused<<<BATCH / 16, 128, 0, stream>>>(inputs, grid, W, outp);
}
